// C2fPSA_78907139162703
// MI455X (gfx1250) — matrix operand order repaired and checked by reading, not compiled, not run on hardware
//
#include <hip/hip_runtime.h>
#include <hip/hip_bf16.h>
#include <math.h>
#include <stdint.h>

// ---------------------------------------------------------------------------
// Types for CDNA5 WMMA / TDM (gfx1250, wave32)
// ---------------------------------------------------------------------------
typedef __attribute__((ext_vector_type(16))) __bf16 v16bf;
typedef __attribute__((ext_vector_type(8)))  float  v8f;
typedef unsigned int v4u __attribute__((ext_vector_type(4)));
typedef int v8i_t     __attribute__((ext_vector_type(8)));
typedef int v4i_t     __attribute__((ext_vector_type(4)));
typedef unsigned short bf16_t;

union FragBf { v16bf v; bf16_t u[16]; uint4 q[2]; };
union AccF   { v8f   v; float  f[8]; };
union Pack8  { uint4 q; bf16_t s[8]; };

#if __has_builtin(__builtin_amdgcn_tensor_load_to_lds) && __has_builtin(__builtin_amdgcn_s_wait_tensorcnt)
#define HAVE_TDM 1
#else
#define HAVE_TDM 0
#endif

__device__ __forceinline__ bf16_t f2bf(float f) {
  union { float f; unsigned int u; } c; c.f = f;
  unsigned int r = c.u + 0x7FFFu + ((c.u >> 16) & 1u);   // round-to-nearest-even
  return (bf16_t)(r >> 16);
}

// ---------------------------------------------------------------------------
// Problem constants
// ---------------------------------------------------------------------------
#define BATCH 2
#define C1    512
#define QD    4
#define HH    24
#define WW2   24
#define HW    576           // 24*24
#define TOK   2304          // QD*HW (tokens per batch)
#define ECD   128
#define GC    512
#define KD    16

// ---------------------------------------------------------------------------
// Weight prep: w = mod * cos(phase)
// ---------------------------------------------------------------------------
__global__ void prepw_bf16_kernel(const float* __restrict__ m, const float* __restrict__ p,
                                  bf16_t* __restrict__ w, int n) {
  int i = blockIdx.x * blockDim.x + threadIdx.x;
  if (i < n) w[i] = f2bf(m[i] * __cosf(p[i]));
}
__global__ void prepw_f32_kernel(const float* __restrict__ m, const float* __restrict__ p,
                                 float* __restrict__ w, int n) {
  int i = blockIdx.x * blockDim.x + threadIdx.x;
  if (i < n) w[i] = m[i] * __cosf(p[i]);
}

// ---------------------------------------------------------------------------
// Channel-range f32 -> bf16 cast/copy: src[nB][Cs][L] (+c0) -> dst[nB][Cd][L] (+d0)
// ---------------------------------------------------------------------------
__global__ void cast_ch_kernel(const float* __restrict__ src, bf16_t* __restrict__ dst,
                               int nB, int Cs, int Cd, int c0, int d0, int nc, int L) {
  long long total = (long long)nB * nc * L;
  long long idx = (long long)blockIdx.x * blockDim.x + threadIdx.x;
  if (idx >= total) return;
  int i = (int)(idx % L); long long t = idx / L;
  int cc = (int)(t % nc); int b = (int)(t / nc);
  dst[((long long)b * Cd + d0 + cc) * L + i] = f2bf(src[((long long)b * Cs + c0 + cc) * L + i]);
}

// ---------------------------------------------------------------------------
// Batched WMMA GEMM:  C[z] = A(MxK, bf16 row-major) x B[z](KxN, bf16 row stride ldb)
// Requirements (all call sites satisfy them): M%64==0, N%64==0, K%32==0,
// pointers 16B-aligned, ldb/ldc multiples of 8.
// 64x64 block tile, 4 waves (2x2 of 16x16 WMMA each), K step 32.
// A tile: DMA'd by the Tensor Data Mover (tensor_load_to_lds) with LDS row
//   padding (16DW interval, 4DW pad) -> fragments are ds_load_b128 pairs.
// B tile: staged transposed (Bt[n][k]) via uint4 global + b16 LDS scatter.
// ---------------------------------------------------------------------------
#define GBM 64
#define GBN 64
#define GBK 32

__global__ __launch_bounds__(128) void gemm_bf16_kernel(
    const bf16_t* __restrict__ A, const bf16_t* __restrict__ Bm,
    float* __restrict__ C, bf16_t* __restrict__ Cbf,
    int M, int N, int K, int ldb, int ldc,
    long long Bbatch, long long Cbatch)
{
  __shared__ __attribute__((aligned(16))) bf16_t As[GBM][GBK + 8];   // 80B rows
  __shared__ __attribute__((aligned(16))) bf16_t Bt[GBN][GBK + 8];   // transposed tile

  const int tid  = threadIdx.x;
  const int lane = tid & 31;
  const int wave = tid >> 5;                 // 0..3
  const int wm   = (wave >> 1) * 32;
  const int wn   = (wave & 1) * 32;
  const int hi   = lane >> 4;
  const int lo   = lane & 15;
  const int bn0  = blockIdx.x * GBN;
  const int bm0  = blockIdx.y * GBM;
  const long long bOff = (long long)blockIdx.z * Bbatch;
  const long long cOff = (long long)blockIdx.z * Cbatch;

  // per-thread staging coordinates (straight-line, no guards)
  const int bkr = tid >> 3;                  // 0..15  B k-row
  const int bn8 = (tid & 7) * 8;             // 0..56  B n-chunk
#if !HAVE_TDM
  const int ar  = tid >> 2;                  // 0..31  A row
  const int ac8 = (tid & 3) * 8;             // 0..24  A k-chunk
#endif

  AccF acc[2][2];
  #pragma unroll
  for (int i = 0; i < 2; i++)
    #pragma unroll
    for (int j = 0; j < 2; j++)
      #pragma unroll
      for (int r = 0; r < 8; r++) acc[i][j].f[r] = 0.f;

  for (int k0 = 0; k0 < K; k0 += GBK) {
    __syncthreads();                         // prev-iter LDS reads done

#if HAVE_TDM
    // ---- A tile via Tensor Data Mover: 64 rows x 64B, row stride K*2B,
    //      LDS pad 4 DWORDs every 16 DWORDs (=> 80B LDS rows).
    if (tid == 0) {
      unsigned lds0 = (unsigned)(uintptr_t)(&As[0][0]);
      unsigned long long ga =
          (unsigned long long)(uintptr_t)(A + (long long)bm0 * K + k0);
      v4u g0 = { 1u,                                   // count=1, user desc
                 lds0,                                 // lds_addr
                 (unsigned)(ga & 0xffffffffull),
                 (unsigned)((ga >> 32) & 0x01ffffffull) | (2u << 30) };  // type=2
      v8i_t g1 = { (int)((1u << 16)                     // data_size = 2B
                       | (1u << 20)                     // pad_enable
                       | (3u << 22)                     // pad_interval: 16 DW
                       | (3u << 25)),                   // pad_amount: 4 DW
                   (int)(32u << 16),                    // tensor_dim0 = 32
                   (int)(64u << 16),                    // tensor_dim1 = 64
                   (int)(32u << 16),                    // tile_dim0 = 32
                   64,                                  // tile_dim1 = 64
                   K,                                   // tensor_dim0_stride
                   0, 0 };
      v4i_t g2 = { 0, 0, 0, 0 };
      v4i_t g3 = { 0, 0, 0, 0 };
      v8i_t g4 = { 0, 0, 0, 0, 0, 0, 0, 0 };
      __builtin_amdgcn_tensor_load_to_lds(g0, g1, g2, g3, g4, 0);
    }
#else
    // ---- A tile manual: 2x uint4 per thread, both loads in flight
    uint4 a0 = *(const uint4*)&A[(long long)(bm0 + ar) * K + k0 + ac8];
    uint4 a1 = *(const uint4*)&A[(long long)(bm0 + ar + 32) * K + k0 + ac8];
    *(uint4*)&As[ar][ac8]      = a0;
    *(uint4*)&As[ar + 32][ac8] = a1;
#endif

    // ---- B tile transposed: both uint4 loads issued before LDS scatter
    Pack8 d0, d1;
    d0.q = *(const uint4*)&Bm[bOff + (long long)(k0 + bkr) * ldb + bn0 + bn8];
    d1.q = *(const uint4*)&Bm[bOff + (long long)(k0 + bkr + 16) * ldb + bn0 + bn8];
    #pragma unroll
    for (int j = 0; j < 8; j++) Bt[bn8 + j][bkr] = d0.s[j];
    #pragma unroll
    for (int j = 0; j < 8; j++) Bt[bn8 + j][bkr + 16] = d1.s[j];

    // prefetch next-iteration tiles (global_prefetch_b8)
    if (k0 + GBK < K) {
      __builtin_prefetch(&A[(long long)(bm0 + (tid >> 2)) * K + k0 + GBK + (tid & 3) * 8], 0, 0);
      __builtin_prefetch(&Bm[bOff + (long long)(k0 + GBK + bkr) * ldb + bn0 + bn8], 0, 0);
    }

#if HAVE_TDM
    __builtin_amdgcn_s_wait_tensorcnt(0);    // TDM write to LDS complete (wave 0)
#endif
    __syncthreads();

    // ---- fragments: all contiguous 16B LDS reads
    FragBf afr[2], bfr[2];
    #pragma unroll
    for (int i = 0; i < 2; i++) {
      int m = wm + i * 16 + lo;
      int kb = hi * 8;
      afr[i].q[0] = *(const uint4*)&As[m][kb];
      afr[i].q[1] = *(const uint4*)&As[m][16 + kb];
    }
    #pragma unroll
    for (int j2 = 0; j2 < 2; j2++) {
      int n = wn + j2 * 16 + lo;
      int kb16 = hi * 16;
      bfr[j2].q[0] = *(const uint4*)&Bt[n][hi * 8];
      bfr[j2].q[1] = *(const uint4*)&Bt[n][16 + hi * 8];
    }
    #pragma unroll
    for (int i = 0; i < 2; i++)
      #pragma unroll
      for (int j2 = 0; j2 < 2; j2++)
        acc[i][j2].v = __builtin_amdgcn_wmma_f32_16x16x32_bf16(
            false, afr[i].v, false, bfr[j2].v, (short)0, acc[i][j2].v, false, false);
  }

  #pragma unroll
  for (int i = 0; i < 2; i++)
    #pragma unroll
    for (int j2 = 0; j2 < 2; j2++)
      #pragma unroll
      for (int r = 0; r < 8; r++) {
        int gm = bm0 + wm + i * 16 + hi * 8 + r;
        int gn = bn0 + wn + j2 * 16 + lo;
        float vv = acc[i][j2].f[r];
        long long o = cOff + (long long)gm * ldc + gn;
        if (C)   C[o]   = vv;
        if (Cbf) Cbf[o] = f2bf(vv);
      }
}

// ---------------------------------------------------------------------------
// Per-channel BN over (batch, spatial): X[nB][C][L], normalize in place,
// optional residual add (after normalize), optional ReLU, optional bf16 copy.
// ---------------------------------------------------------------------------
__global__ __launch_bounds__(256) void bn_kernel(
    float* __restrict__ X, const float* __restrict__ Res, bf16_t* __restrict__ Obf,
    int nB, int C, int L, int relu)
{
  int ch = blockIdx.x;
  __shared__ float ssum[256], ssq[256];
  float s = 0.f, q = 0.f;
  for (int b = 0; b < nB; b++) {
    long long base = ((long long)b * C + ch) * L;
    for (int i = threadIdx.x; i < L; i += 256) { float v = X[base + i]; s += v; q += v * v; }
  }
  ssum[threadIdx.x] = s; ssq[threadIdx.x] = q; __syncthreads();
  for (int off = 128; off > 0; off >>= 1) {
    if (threadIdx.x < off) { ssum[threadIdx.x] += ssum[threadIdx.x + off];
                             ssq[threadIdx.x]  += ssq[threadIdx.x + off]; }
    __syncthreads();
  }
  float cnt  = (float)nB * (float)L;
  float mean = ssum[0] / cnt;
  float var  = ssq[0] / cnt - mean * mean;
  float rinv = rsqrtf(var + 1e-5f);
  for (int b = 0; b < nB; b++) {
    long long base = ((long long)b * C + ch) * L;
    for (int i = threadIdx.x; i < L; i += 256) {
      float v = (X[base + i] - mean) * rinv;
      if (Res)  v += Res[base + i];
      if (relu) v = fmaxf(v, 0.f);
      X[base + i] = v;
      if (Obf) Obf[base + i] = f2bf(v);
    }
  }
}

// ---------------------------------------------------------------------------
// Fused multi-head attention. One block = one (b,h,q) slice x 16 query tokens.
// qkv: bf16 [B][768][TOK] (q: ch 0..255, k: 256..511, v: 512..767)
// outbf: bf16 [B][256][TOK]
// ---------------------------------------------------------------------------
__global__ __launch_bounds__(128) void attn_kernel(
    const bf16_t* __restrict__ qkv, bf16_t* __restrict__ outbf, float scale)
{
  __shared__ __attribute__((aligned(16))) float  Sf[16][HW];     // 36864 B
  __shared__ __attribute__((aligned(16))) bf16_t Pbf[16][HW];    // 18432 B
  __shared__ __attribute__((aligned(16))) bf16_t kvs[16 * HW];   // 18432 B (K^T then V)
  __shared__ __attribute__((aligned(16))) bf16_t qs[16 * 16];
  __shared__ float red[16][8];
  __shared__ float outp[4][256];

  const int z  = blockIdx.y;                 // 0..127  (b,h,q)
  const int nb = blockIdx.x;                 // 0..35   query block
  const int b  = z >> 6;
  const int h  = (z >> 2) & 15;
  const int q  = z & 3;
  const int tid = threadIdx.x, lane = tid & 31, wave = tid >> 5;
  const int hi = lane >> 4, lo = lane & 15;
  const int n0 = nb * 16;
  const uint4 zero4 = make_uint4(0u, 0u, 0u, 0u);

  const long long base = (long long)b * 768 * TOK;
  const bf16_t* Qp = qkv + base + (long long)(h * KD) * TOK + q * HW;
  const bf16_t* Kp = qkv + base + (long long)(256 + h * KD) * TOK + q * HW;
  const bf16_t* Vp = qkv + base + (long long)(512 + h * KD) * TOK + q * HW;

  // stage Q^T (16x16) and K^T (576x16)
  #pragma unroll
  for (int e = tid; e < 256; e += 128) {
    int c = e >> 4, n = e & 15;
    qs[n * 16 + c] = Qp[(long long)c * TOK + n0 + n];
  }
  #pragma unroll
  for (int v = tid; v < (16 * HW) / 8; v += 128) {    // 1152 vec8, 9 per thread
    int c = v / 72;
    int t8 = (v % 72) * 8;
    Pack8 d; d.q = *(const uint4*)&Kp[(long long)c * TOK + t8];
    #pragma unroll
    for (int j = 0; j < 8; j++) kvs[(t8 + j) * 16 + c] = d.s[j];
  }
  __syncthreads();

  // A fragment from Q (16 queries x K=16 channels, zero-padded to 32)
  FragBf aq;
  aq.q[0] = *(const uint4*)&qs[lo * 16 + hi * 8];
  aq.q[1] = zero4;

  // ---- phase 1: scores S = scale * Q^T K ----
  for (int i = 0; i < 9; i++) {
    int mt = (wave * 9 + i) * 16;
    FragBf bk;
    bk.q[0] = *(const uint4*)&kvs[(mt + lo) * 16 + hi * 8];
    bk.q[1] = zero4;
    AccF acc;
    #pragma unroll
    for (int r = 0; r < 8; r++) acc.f[r] = 0.f;
    acc.v = __builtin_amdgcn_wmma_f32_16x16x32_bf16(
        false, aq.v, false, bk.v, (short)0, acc.v, false, false);
    #pragma unroll
    for (int r = 0; r < 8; r++) Sf[hi * 8 + r][mt + lo] = acc.f[r] * scale;
  }
  __syncthreads();

  // stage V (16x576, row-major) into kvs — K^T is dead now
  #pragma unroll
  for (int v = tid; v < (16 * HW) / 8; v += 128) {
    int d = v / 72;
    int t8 = (v % 72) * 8;
    *(uint4*)&kvs[d * HW + t8] = *(const uint4*)&Vp[(long long)d * TOK + t8];
  }

  // ---- phase 2: row softmax (fp32), write P as bf16 ----
  int row = tid >> 3, sub = tid & 7;
  float mx = -3.402823e38f;
  for (int m = sub; m < HW; m += 8) mx = fmaxf(mx, Sf[row][m]);
  red[row][sub] = mx; __syncthreads();
  if (sub == 0) { float v = red[row][0];
                  for (int j2 = 1; j2 < 8; j2++) v = fmaxf(v, red[row][j2]);
                  red[row][0] = v; }
  __syncthreads();
  mx = red[row][0];
  float sm = 0.f;
  for (int m = sub; m < HW; m += 8) { float e = __expf(Sf[row][m] - mx); Sf[row][m] = e; sm += e; }
  __syncthreads();
  red[row][sub] = sm; __syncthreads();
  if (sub == 0) { float v = 0.f; for (int j2 = 0; j2 < 8; j2++) v += red[row][j2]; red[row][0] = v; }
  __syncthreads();
  float inv = 1.f / red[row][0];
  for (int m = sub; m < HW; m += 8) Pbf[row][m] = f2bf(Sf[row][m] * inv);
  __syncthreads();   // also covers V staging vs phase-3 reads

  // ---- phase 3: out(16n x 16d) = sum_m P[n][m] * V[d][m], K chunks of 32 ----
  AccF oacc;
  #pragma unroll
  for (int r = 0; r < 8; r++) oacc.f[r] = 0.f;
  for (int kc = wave; kc < 18; kc += 4) {
    FragBf ap, bv;
    int kb = hi * 8;
    ap.q[0] = *(const uint4*)&Pbf[lo][kc * 32 + kb];
    ap.q[1] = *(const uint4*)&Pbf[lo][kc * 32 + 16 + kb];
    int kb16 = hi * 16;
    bv.q[0] = *(const uint4*)&kvs[lo * HW + kc * 32 + kb];
    bv.q[1] = *(const uint4*)&kvs[lo * HW + kc * 32 + 16 + kb];
    oacc.v = __builtin_amdgcn_wmma_f32_16x16x32_bf16(
        false, ap.v, false, bv.v, (short)0, oacc.v, false, false);
  }
  #pragma unroll
  for (int r = 0; r < 8; r++) outp[wave][(hi * 8 + r) * 16 + lo] = oacc.f[r];
  __syncthreads();
  for (int t = tid; t < 256; t += 128) {
    float s = outp[0][t] + outp[1][t] + outp[2][t] + outp[3][t];
    int nr = t >> 4, d = t & 15;
    outbf[(long long)b * 256 * TOK + (long long)(h * KD + d) * TOK + q * HW + n0 + nr] = f2bf(s);
  }
}

// ---------------------------------------------------------------------------
// pe depthwise 3x3 conv on b-half + attn-proj + shortcut:
// abuf = aproj_out + dwconv3x3(b) + b ; also bf16 copy.
// ---------------------------------------------------------------------------
__global__ void pe_add_kernel(const float* __restrict__ ynorm, const float* __restrict__ wpe,
                              const float* __restrict__ ao, float* __restrict__ abuf,
                              bf16_t* __restrict__ abuf_bf, int total)
{
  int idx = blockIdx.x * blockDim.x + threadIdx.x;
  if (idx >= total) return;
  int t = idx % TOK; int tmp = idx / TOK; int cc = tmp % 256; int b = tmp / 256;
  int q = t / HW, hw = t % HW, hh = hw / WW2, ww = hw % WW2;
  const float* bin = ynorm + ((long long)b * C1 + 256 + cc) * TOK + q * HW;
  float accv = 0.f;
  #pragma unroll
  for (int ky = 0; ky < 3; ky++) {
    int y2 = hh + ky - 1; if (y2 < 0 || y2 >= HH) continue;
    #pragma unroll
    for (int kx = 0; kx < 3; kx++) {
      int x2 = ww + kx - 1; if (x2 < 0 || x2 >= WW2) continue;
      accv += wpe[cc * 9 + ky * 3 + kx] * bin[y2 * WW2 + x2];
    }
  }
  float v = ao[idx] + accv + bin[hw];
  abuf[idx] = v; abuf_bf[idx] = f2bf(v);
}

// ---------------------------------------------------------------------------
// g = guide @ gl_w^T + gl_b  ->  g[B][ECD]
// ---------------------------------------------------------------------------
__global__ void glin_kernel(const float* __restrict__ guide, const float* __restrict__ glw,
                            const float* __restrict__ glb, float* __restrict__ g)
{
  int idx = blockIdx.x * blockDim.x + threadIdx.x;
  if (idx >= BATCH * ECD) return;
  int b = idx / ECD, e = idx % ECD;
  float s = glb[e];
  for (int i = 0; i < GC; i++) s += guide[b * GC + i] * glw[e * GC + i];
  g[idx] = s;
}

// ---------------------------------------------------------------------------
// gate[b][q] = sigmoid( sum_{c,hw} e[b][c][q*HW+hw] * g[b][c] / sqrt(ECD*HW) )
// ---------------------------------------------------------------------------
__global__ __launch_bounds__(256) void gate_kernel(const float* __restrict__ e,
                                                   const float* __restrict__ g,
                                                   float* __restrict__ gate)
{
  int b = blockIdx.x >> 2, q = blockIdx.x & 3;
  __shared__ float sb[256];
  float s = 0.f;
  for (int i = threadIdx.x; i < ECD * HW; i += 256) {
    int cc = i / HW, hw = i % HW;
    s += e[((long long)b * ECD + cc) * TOK + q * HW + hw] * g[b * ECD + cc];
  }
  sb[threadIdx.x] = s; __syncthreads();
  for (int off = 128; off > 0; off >>= 1) {
    if (threadIdx.x < off) sb[threadIdx.x] += sb[threadIdx.x + off];
    __syncthreads();
  }
  if (threadIdx.x == 0) {
    float a = sb[0] / sqrtf((float)(ECD * HW));
    gate[blockIdx.x] = 1.f / (1.f + __expf(-a));
  }
}

// ---------------------------------------------------------------------------
// im2col for mproj (3x3 pad 1) with gate scaling
// ---------------------------------------------------------------------------
__global__ void im2col_kernel(const float* __restrict__ e, const float* __restrict__ gate,
                              bf16_t* __restrict__ col, int total)
{
  int idx = blockIdx.x * blockDim.x + threadIdx.x;
  if (idx >= total) return;
  int hw = idx % HW; int tmp = idx / HW; int k = tmp % (ECD * 9); int img = tmp / (ECD * 9);
  int b = img >> 2, q = img & 3;
  int cc = k / 9, kk = k % 9, ky = kk / 3, kx = kk % 3;
  int hh = hw / WW2 + ky - 1, ww = hw % WW2 + kx - 1;
  float v = 0.f;
  if (hh >= 0 && hh < HH && ww >= 0 && ww < WW2)
    v = e[((long long)b * ECD + cc) * TOK + q * HW + hh * WW2 + ww] * gate[b * 4 + q];
  col[idx] = f2bf(v);
}

// ---------------------------------------------------------------------------
// mtmp[8 imgs][256][HW] -> cat channels 768..1023 of cat_bf[B][1024][TOK]
// ---------------------------------------------------------------------------
__global__ void mcat_kernel(const float* __restrict__ mtmp, bf16_t* __restrict__ cat)
{
  int idx = blockIdx.x * blockDim.x + threadIdx.x;
  int total = BATCH * 256 * TOK;
  if (idx >= total) return;
  int hw = idx % HW; int tmp = idx / HW; int q = tmp % 4; tmp /= 4;
  int o = tmp % 256; int b = tmp / 256;
  cat[((long long)b * 1024 + 768 + o) * TOK + q * HW + hw] =
      f2bf(mtmp[((long long)(b * 4 + q) * 256 + o) * HW + hw]);
}

// ---------------------------------------------------------------------------
// Orchestration
// ---------------------------------------------------------------------------
extern "C" void kernel_launch(void* const* d_in, const int* in_sizes, int n_in,
                              void* d_out, int out_size, void* d_ws, size_t ws_size,
                              hipStream_t stream)
{
  (void)in_sizes; (void)n_in; (void)out_size; (void)ws_size;
  const float* x      = (const float*)d_in[0];
  const float* guide  = (const float*)d_in[1];
  const float* cv1_m  = (const float*)d_in[2];
  const float* cv1_p  = (const float*)d_in[3];
  const float* qkv_m  = (const float*)d_in[4];
  const float* qkv_p  = (const float*)d_in[5];
  const float* apr_m  = (const float*)d_in[6];
  const float* apr_p  = (const float*)d_in[7];
  const float* pe_m   = (const float*)d_in[8];
  const float* pe_p   = (const float*)d_in[9];
  const float* f1_m   = (const float*)d_in[10];
  const float* f1_p   = (const float*)d_in[11];
  const float* f2_m   = (const float*)d_in[12];
  const float* f2_p   = (const float*)d_in[13];
  const float* ec_m   = (const float*)d_in[14];
  const float* ec_p   = (const float*)d_in[15];
  const float* gl_w   = (const float*)d_in[16];
  const float* gl_b   = (const float*)d_in[17];
  const float* mp_m   = (const float*)d_in[18];
  const float* mp_p   = (const float*)d_in[19];
  const float* cv2_m  = (const float*)d_in[20];
  const float* cv2_p  = (const float*)d_in[21];
  float* out = (float*)d_out;

  char* ws = (char*)d_ws;
  size_t off = 0;
  auto alloc = [&](size_t bytes) -> void* {
    void* p = ws + off;
    off = (off + bytes + 255) & ~(size_t)255;
    return p;
  };

  bf16_t* w_cv1 = (bf16_t*)alloc((size_t)512 * 512 * 2);
  bf16_t* w_qkv = (bf16_t*)alloc((size_t)768 * 256 * 2);
  bf16_t* w_apr = (bf16_t*)alloc((size_t)256 * 256 * 2);
  float*  w_pe  = (float*) alloc((size_t)256 * 9 * 4);
  bf16_t* w_f1  = (bf16_t*)alloc((size_t)512 * 256 * 2);
  bf16_t* w_f2  = (bf16_t*)alloc((size_t)256 * 512 * 2);
  bf16_t* w_ec  = (bf16_t*)alloc((size_t)128 * 256 * 2);
  bf16_t* w_mp  = (bf16_t*)alloc((size_t)256 * 1152 * 2);
  bf16_t* w_cv2 = (bf16_t*)alloc((size_t)512 * 1024 * 2);

  bf16_t* x_bf    = (bf16_t*)alloc((size_t)BATCH * 512 * TOK * 2);
  float*  y       = (float*) alloc((size_t)BATCH * 512 * TOK * 4);
  bf16_t* b_bf    = (bf16_t*)alloc((size_t)BATCH * 256 * TOK * 2);
  bf16_t* qkv_bf  = (bf16_t*)alloc((size_t)BATCH * 768 * TOK * 2);
  bf16_t* attn_bf = (bf16_t*)alloc((size_t)BATCH * 256 * TOK * 2);
  float*  ao      = (float*) alloc((size_t)BATCH * 256 * TOK * 4);
  float*  abuf    = (float*) alloc((size_t)BATCH * 256 * TOK * 4);
  bf16_t* abuf_bf = (bf16_t*)alloc((size_t)BATCH * 256 * TOK * 2);
  float*  hb      = (float*) alloc((size_t)BATCH * 512 * TOK * 4);
  bf16_t* h_bf    = (bf16_t*)alloc((size_t)BATCH * 512 * TOK * 2);
  float*  fb      = (float*) alloc((size_t)BATCH * 256 * TOK * 4);  // becomes p (f32)
  bf16_t* p_bf    = (bf16_t*)alloc((size_t)BATCH * 256 * TOK * 2);
  float*  eb      = (float*) alloc((size_t)BATCH * 128 * TOK * 4);
  float*  gbuf    = (float*) alloc((size_t)BATCH * ECD * 4);
  float*  gatev   = (float*) alloc((size_t)BATCH * 4 * 4);
  bf16_t* col     = (bf16_t*)alloc((size_t)8 * 1152 * HW * 2);
  float*  mtmp    = (float*) alloc((size_t)8 * 256 * HW * 4);
  bf16_t* cat_bf  = (bf16_t*)alloc((size_t)BATCH * 1024 * TOK * 2);

  auto prepw = [&](const float* m, const float* p, bf16_t* w, int n) {
    prepw_bf16_kernel<<<(n + 255) / 256, 256, 0, stream>>>(m, p, w, n);
  };
  prepw(cv1_m, cv1_p, w_cv1, 512 * 512);
  prepw(qkv_m, qkv_p, w_qkv, 768 * 256);
  prepw(apr_m, apr_p, w_apr, 256 * 256);
  prepw(f1_m,  f1_p,  w_f1,  512 * 256);
  prepw(f2_m,  f2_p,  w_f2,  256 * 512);
  prepw(ec_m,  ec_p,  w_ec,  128 * 256);
  prepw(mp_m,  mp_p,  w_mp,  256 * 1152);
  prepw(cv2_m, cv2_p, w_cv2, 512 * 1024);
  prepw_f32_kernel<<<(256 * 9 + 255) / 256, 256, 0, stream>>>(pe_m, pe_p, w_pe, 256 * 9);

  auto gemm = [&](const bf16_t* A, const bf16_t* Bm, float* C, bf16_t* Cbf,
                  int M, int N, int K, int ldb, int ldc, long long Bb, long long Cb, int nz) {
    dim3 g((N + GBN - 1) / GBN, (M + GBM - 1) / GBM, nz);
    gemm_bf16_kernel<<<g, 128, 0, stream>>>(A, Bm, C, Cbf, M, N, K, ldb, ldc, Bb, Cb);
  };
  auto bn = [&](float* X, const float* Res, bf16_t* Obf, int nB, int C, int L, int relu) {
    bn_kernel<<<C, 256, 0, stream>>>(X, Res, Obf, nB, C, L, relu);
  };
  auto castch = [&](const float* src, bf16_t* dst, int nB, int Cs, int Cd,
                    int c0, int d0, int nc, int L) {
    long long total = (long long)nB * nc * L;
    cast_ch_kernel<<<(unsigned)((total + 255) / 256), 256, 0, stream>>>(
        src, dst, nB, Cs, Cd, c0, d0, nc, L);
  };

  // --- stage 1: cv1 + BN + ReLU ---
  castch(x, x_bf, BATCH, 512, 512, 0, 0, 512, TOK);
  gemm(w_cv1, x_bf, y, nullptr, 512, TOK, 512, TOK, TOK,
       (long long)512 * TOK, (long long)512 * TOK, BATCH);
  bn(y, nullptr, nullptr, BATCH, 512, TOK, 1);
  castch(y, b_bf, BATCH, 512, 256, 256, 0, 256, TOK);   // b half -> bf16

  // --- PSA: qkv GEMM -> fused attention -> aproj -> pe + shortcut ---
  gemm(w_qkv, b_bf, nullptr, qkv_bf, 768, TOK, 256, TOK, TOK,
       (long long)256 * TOK, (long long)768 * TOK, BATCH);
  {
    dim3 g(36, 128);
    attn_kernel<<<g, 128, 0, stream>>>(qkv_bf, attn_bf, 0.25f);  // kd^-0.5
  }
  gemm(w_apr, attn_bf, ao, nullptr, 256, TOK, 256, TOK, TOK,
       (long long)256 * TOK, (long long)256 * TOK, BATCH);
  {
    int total = BATCH * 256 * TOK;
    pe_add_kernel<<<(total + 255) / 256, 256, 0, stream>>>(y, w_pe, ao, abuf, abuf_bf, total);
  }

  // --- FFN: ffn1 -> BN+ReLU -> ffn2 -> BN + residual => p ---
  gemm(w_f1, abuf_bf, hb, nullptr, 512, TOK, 256, TOK, TOK,
       (long long)256 * TOK, (long long)512 * TOK, BATCH);
  bn(hb, nullptr, h_bf, BATCH, 512, TOK, 1);
  gemm(w_f2, h_bf, fb, nullptr, 256, TOK, 512, TOK, TOK,
       (long long)512 * TOK, (long long)256 * TOK, BATCH);
  bn(fb, abuf, p_bf, BATCH, 256, TOK, 0);               // p = bn(f) + abuf

  // --- MSAB: ec -> BN+ReLU -> gate -> im2col -> mproj -> BN+ReLU ---
  gemm(w_ec, p_bf, eb, nullptr, 128, TOK, 256, TOK, TOK,
       (long long)256 * TOK, (long long)128 * TOK, BATCH);
  bn(eb, nullptr, nullptr, BATCH, 128, TOK, 1);
  glin_kernel<<<1, 256, 0, stream>>>(guide, gl_w, gl_b, gbuf);
  gate_kernel<<<8, 256, 0, stream>>>(eb, gbuf, gatev);
  {
    int total = 8 * 1152 * HW;
    im2col_kernel<<<(total + 255) / 256, 256, 0, stream>>>(eb, gatev, col, total);
  }
  gemm(w_mp, col, mtmp, nullptr, 256, HW, 1152, HW, HW,
       (long long)1152 * HW, (long long)256 * HW, 8);
  bn(mtmp, nullptr, nullptr, 8, 256, HW, 1);

  // --- concat [a, b, p, m] -> cv2 -> BN+ReLU -> out ---
  castch(y,  cat_bf, BATCH, 512, 1024, 0,   0,   256, TOK);   // a
  castch(y,  cat_bf, BATCH, 512, 1024, 256, 256, 256, TOK);   // b
  castch(fb, cat_bf, BATCH, 256, 1024, 0,   512, 256, TOK);   // p
  {
    int total = BATCH * 256 * TOK;
    mcat_kernel<<<(total + 255) / 256, 256, 0, stream>>>(mtmp, cat_bf);
  }
  gemm(w_cv2, cat_bf, out, nullptr, 512, TOK, 1024, TOK, TOK,
       (long long)1024 * TOK, (long long)512 * TOK, BATCH);
  bn(out, nullptr, nullptr, BATCH, 512, TOK, 1);
}
